// GRUEncoder_33303176413505
// MI455X (gfx1250) — matrix operand order repaired and checked by reading, not compiled, not run on hardware
//
#include <hip/hip_runtime.h>
#include <hip/hip_bf16.h>

typedef __attribute__((ext_vector_type(8)))  _Float16 v8h;
typedef __attribute__((ext_vector_type(16))) _Float16 v16h;
typedef __attribute__((ext_vector_type(8)))  float    v8f;

#define BATCH 32
#define TLEN  8192
#define HID   64
#define G3    192     // 3*H
#define LAT   128

// ---------- fast transcendentals (single v_exp_f32 / v_rcp_f32) ----------
__device__ __forceinline__ float fast_sigmoid(float x) {
    float e = __builtin_amdgcn_exp2f(-1.4426950408889634f * x);
    return __builtin_amdgcn_rcpf(1.0f + e);
}
__device__ __forceinline__ float fast_tanh(float x) {
    float e = __builtin_amdgcn_exp2f(-2.8853900817779268f * x);
    return 2.0f * __builtin_amdgcn_rcpf(1.0f + e) - 1.0f;
}

// ---------- tile loaders (ISA 7.12.2 layouts) ----------
// combine two 8-half chunks into a v16h
__device__ __forceinline__ v16h ld16(const _Float16* p0, const _Float16* p1) {
    v8h lo = *(const v8h*)p0;
    v8h hi = *(const v8h*)p1;
    v16h r;
#pragma unroll
    for (int i = 0; i < 8; ++i) { r[i] = lo[i]; r[i + 8] = hi[i]; }
    return r;
}
// A 16x32 f16 tile, row-major source with leading-dim ld (elements).
// lanes 0-15: elems 0-7 -> K 0..7, elems 8-15 -> K 16..23 ; lanes 16-31: +8.
__device__ __forceinline__ v16h ldA(const _Float16* base, int row, int ld,
                                    int kbase, int hiHalf) {
    const _Float16* p = base + (size_t)row * ld + kbase + hiHalf * 8;
    return ld16(p, p + 16);
}
// B 32x16 f16 tile: column n contiguous over K.
// lanes 0-15: elems 0-7 -> K 0..7, elems 8-15 -> K 16..23 ; lanes 16-31: +8.
__device__ __forceinline__ v16h ldB(const _Float16* wrow, int kbase, int hiHalf) {
    const _Float16* p = wrow + kbase + hiHalf * 8;
    return ld16(p, p + 16);
}
__device__ __forceinline__ v8f wmma_f16(v16h a, v16h b, v8f c) {
    return __builtin_amdgcn_wmma_f32_16x16x32_f16(false, a, false, b, (short)0, c,
                                                  false, false);
}

// ---------- kernel 1: f32 -> f16 weight convert ----------
__global__ void k_cvt_f16(const float* __restrict__ src, _Float16* __restrict__ dst,
                          int n) {
    int i = blockIdx.x * blockDim.x + threadIdx.x;
    if (i < n) dst[i] = (_Float16)src[i];
}

// ---------- kernel 2: input projection x = tanh(W_in @ [f0,loud] + b_in) ----------
__global__ void k_inproj(const float* __restrict__ f0, const float* __restrict__ lo,
                         const float* __restrict__ Win, const float* __restrict__ bin,
                         _Float16* __restrict__ x, int n) {
    int i = blockIdx.x * blockDim.x + threadIdx.x;
    if (i >= n) return;
    int row = i >> 6, j = i & 63;
    float v = Win[j * 2] * f0[row] + Win[j * 2 + 1] * lo[row] + bin[j];
    x[i] = (_Float16)fast_tanh(v);
}

// ---------- kernel 3: xg = x @ Wih^T + bih  (M=B*T, N=192, K=64) ----------
// out layout (T, B, 192) so the scan reads one contiguous slab per step.
__global__ __launch_bounds__(128) void k_xg_gemm(
    const _Float16* __restrict__ xin,   // (B*T, 64) row-major
    const _Float16* __restrict__ Wih,   // (192, 64) f16, this layer
    const float* __restrict__ bih,      // (192)
    _Float16* __restrict__ xg)          // (T, 32, 192)
{
    int tid = threadIdx.x, wv = tid >> 5, ln = tid & 31;
    int lane16 = ln & 15, hiHalf = (ln >> 4) & 1;
    long r0 = (long)blockIdx.x * 16;

    v16h Bt[3][2]; float bias[3];
#pragma unroll
    for (int ta = 0; ta < 3; ++ta) {
        int g = 48 * wv + 16 * ta + lane16;
        bias[ta] = bih[g];
#pragma unroll
        for (int k = 0; k < 2; ++k) Bt[ta][k] = ldB(Wih + g * 64, k * 32, hiHalf);
    }
    v16h A[2];
#pragma unroll
    for (int k = 0; k < 2; ++k) A[k] = ldA(xin, (int)(r0 + lane16), 64, k * 32, hiHalf);

    v8f acc[3] = {};
#pragma unroll
    for (int ta = 0; ta < 3; ++ta) {
        acc[ta] = wmma_f16(A[0], Bt[ta][0], acc[ta]);
        acc[ta] = wmma_f16(A[1], Bt[ta][1], acc[ta]);
    }
    int bb = (int)(r0 >> 13);        // row / T
    int tt = (int)(r0 & (TLEN - 1)); // row % T
#pragma unroll
    for (int ta = 0; ta < 3; ++ta) {
        int g = 48 * wv + 16 * ta + lane16;
#pragma unroll
        for (int v = 0; v < 8; ++v) {
            int m = v + hiHalf * 8;
            xg[((size_t)(tt + m) * BATCH + bb) * G3 + g] =
                (_Float16)(acc[ta][v] + bias[ta]);
        }
    }
}

// ---------- kernel 4: GRU recurrence (sequential over T, 16 batch rows / block) ----------
__global__ __launch_bounds__(128) void k_gru_scan(
    const _Float16* __restrict__ xg,    // (T, 32, 192)
    const _Float16* __restrict__ Whh,   // (192, 64) f16, this layer
    const float* __restrict__ bhh,      // (192)
    _Float16* __restrict__ y)           // (32, T, 64)
{
    __shared__ _Float16 h_sm[16 * 64];  // current h, row-major
    __shared__ float hg_sm[16 * G3];    // h @ Whh^T staging

    int tid = threadIdx.x, wv = tid >> 5, ln = tid & 31;
    int lane16 = ln & 15, hiHalf = (ln >> 4) & 1;
    int b0 = blockIdx.x * 16;

    // Whh resident in VGPRs for all 8192 steps
    v16h Bt[3][2];
#pragma unroll
    for (int ta = 0; ta < 3; ++ta) {
        int g = 48 * wv + 16 * ta + lane16;
#pragma unroll
        for (int k = 0; k < 2; ++k) Bt[ta][k] = ldB(Whh + g * 64, k * 32, hiHalf);
    }
    // gate-phase mapping: thread owns (m=gm, j0..j0+7)
    int gm = tid & 15, j0 = (tid >> 4) * 8;
    float br[8], bz[8], bn[8];
#pragma unroll
    for (int k = 0; k < 8; ++k) {
        br[k] = bhh[j0 + k];
        bz[k] = bhh[64 + j0 + k];
        bn[k] = bhh[128 + j0 + k];
    }
    { v8h z = {}; *(v8h*)(h_sm + tid * 8) = z; }  // h0 = 0
    __syncthreads();

    for (int t = 0; t < TLEN; ++t) {
        // prefetch this step's xg early (hidden under WMMA phase)
        const _Float16* xp = xg + ((size_t)t * BATCH + (b0 + gm)) * G3 + j0;
        v8h xr = *(const v8h*)xp;
        v8h xz = *(const v8h*)(xp + 64);
        v8h xn = *(const v8h*)(xp + 128);

        // h @ Whh^T via WMMA (A relayout from LDS)
        v16h A[2];
#pragma unroll
        for (int k = 0; k < 2; ++k) A[k] = ldA(h_sm, lane16, 64, k * 32, hiHalf);
        v8f acc[3] = {};
#pragma unroll
        for (int ta = 0; ta < 3; ++ta) {
            acc[ta] = wmma_f16(A[0], Bt[ta][0], acc[ta]);
            acc[ta] = wmma_f16(A[1], Bt[ta][1], acc[ta]);
        }
#pragma unroll
        for (int ta = 0; ta < 3; ++ta) {
            int g = 48 * wv + 16 * ta + lane16;
#pragma unroll
            for (int v = 0; v < 8; ++v) hg_sm[(v + hiHalf * 8) * G3 + g] = acc[ta][v];
        }
        __syncthreads();

        // gate phase (f32)
        v8h hold = *(const v8h*)(h_sm + gm * 64 + j0);
        const float* hgr = hg_sm + gm * G3;
        v8h hnew;
#pragma unroll
        for (int k = 0; k < 8; ++k) {
            float hrv = hgr[j0 + k] + br[k];
            float hzv = hgr[64 + j0 + k] + bz[k];
            float hnv = hgr[128 + j0 + k] + bn[k];
            float r = fast_sigmoid((float)xr[k] + hrv);
            float z = fast_sigmoid((float)xz[k] + hzv);
            float n = fast_tanh((float)xn[k] + r * hnv);
            hnew[k] = (_Float16)((1.0f - z) * n + z * (float)hold[k]);
        }
        *(v8h*)(h_sm + gm * 64 + j0) = hnew;
        *(v8h*)(y + ((size_t)(b0 + gm) * TLEN + t) * 64 + j0) = hnew;
        __syncthreads();
    }
}

// ---------- kernel 5: z = y @ Wout^T + b_out, fused LayerNorm ----------
__global__ __launch_bounds__(128) void k_out_ln(
    const _Float16* __restrict__ yin,   // (B*T, 64)
    const _Float16* __restrict__ Wout,  // (128, 64) f16
    const float* __restrict__ bout, const float* __restrict__ gamma,
    const float* __restrict__ beta, float* __restrict__ out) // (B*T, 128)
{
    __shared__ float zsm[16 * LAT];
    __shared__ float red[16 * 16];
    __shared__ float stats[32];

    int tid = threadIdx.x, wv = tid >> 5, ln = tid & 31;
    int lane16 = ln & 15, hiHalf = (ln >> 4) & 1;
    long r0 = (long)blockIdx.x * 16;

    v16h Bt[2][2]; float bias[2];
#pragma unroll
    for (int ta = 0; ta < 2; ++ta) {
        int n = 32 * wv + 16 * ta + lane16;
        bias[ta] = bout[n];
#pragma unroll
        for (int k = 0; k < 2; ++k) Bt[ta][k] = ldB(Wout + n * 64, k * 32, hiHalf);
    }
    v16h A[2];
#pragma unroll
    for (int k = 0; k < 2; ++k) A[k] = ldA(yin, (int)(r0 + lane16), 64, k * 32, hiHalf);

    v8f acc[2] = {};
#pragma unroll
    for (int ta = 0; ta < 2; ++ta) {
        acc[ta] = wmma_f16(A[0], Bt[ta][0], acc[ta]);
        acc[ta] = wmma_f16(A[1], Bt[ta][1], acc[ta]);
    }
#pragma unroll
    for (int ta = 0; ta < 2; ++ta) {
        int n = 32 * wv + 16 * ta + lane16;
#pragma unroll
        for (int v = 0; v < 8; ++v)
            zsm[(v + hiHalf * 8) * LAT + n] = acc[ta][v] + bias[ta];
    }
    __syncthreads();

    int m = tid >> 3, p = tid & 7;
    const float* zr = zsm + m * LAT + p * 16;
    float s = 0.f, s2 = 0.f;
#pragma unroll
    for (int c = 0; c < 16; ++c) { float v = zr[c]; s += v; s2 += v * v; }
    red[m * 16 + p] = s;
    red[m * 16 + 8 + p] = s2;
    __syncthreads();
    if (tid < 16) {
        float ss = 0.f, ss2 = 0.f;
#pragma unroll
        for (int q = 0; q < 8; ++q) { ss += red[tid * 16 + q]; ss2 += red[tid * 16 + 8 + q]; }
        float mu = ss * (1.0f / 128.0f);
        float var = ss2 * (1.0f / 128.0f) - mu * mu;
        stats[tid * 2] = mu;
        stats[tid * 2 + 1] = __builtin_amdgcn_rsqf(var + 1e-5f);
    }
    __syncthreads();
    float mu = stats[m * 2], rs = stats[m * 2 + 1];
    float* op = out + (size_t)(r0 + m) * LAT + p * 16;
#pragma unroll
    for (int c = 0; c < 16; ++c)
        op[c] = (zr[c] - mu) * rs * gamma[p * 16 + c] + beta[p * 16 + c];
}

// ---------- host ----------
extern "C" void kernel_launch(void* const* d_in, const int* in_sizes, int n_in,
                              void* d_out, int out_size, void* d_ws, size_t ws_size,
                              hipStream_t stream) {
    const float* f0    = (const float*)d_in[0];
    const float* loud  = (const float*)d_in[1];
    const float* W_in  = (const float*)d_in[2];
    const float* b_in  = (const float*)d_in[3];
    const float* W_ih  = (const float*)d_in[4];   // (2,192,64)
    const float* W_hh  = (const float*)d_in[5];   // (2,192,64)
    const float* b_ih  = (const float*)d_in[6];   // (2,192)
    const float* b_hh  = (const float*)d_in[7];   // (2,192)
    const float* W_out = (const float*)d_in[8];   // (128,64)
    const float* b_out = (const float*)d_in[9];
    const float* gamma = (const float*)d_in[10];
    const float* beta  = (const float*)d_in[11];
    float* out = (float*)d_out;

    char* ws = (char*)d_ws;
    const size_t OFF_WIH  = 0;                              // 24576 f16
    const size_t OFF_WHH  = 49152;                          // 24576 f16
    const size_t OFF_WOUT = 98304;                          // 8192 f16
    const size_t OFF_X0   = 114688;                         // 16M f16
    const size_t OFF_Y0   = OFF_X0 + (size_t)BATCH * TLEN * HID * 2;
    const size_t OFF_XG   = OFF_Y0 + (size_t)BATCH * TLEN * HID * 2;

    _Float16* wih_h  = (_Float16*)(ws + OFF_WIH);
    _Float16* whh_h  = (_Float16*)(ws + OFF_WHH);
    _Float16* wout_h = (_Float16*)(ws + OFF_WOUT);
    _Float16* x0     = (_Float16*)(ws + OFF_X0);   // layer0 input, later y of layer1
    _Float16* y0     = (_Float16*)(ws + OFF_Y0);   // layer0 output
    _Float16* xgbuf  = (_Float16*)(ws + OFF_XG);   // (T,32,192) f16

    // weight converts
    k_cvt_f16<<<(2 * G3 * HID + 255) / 256, 256, 0, stream>>>(W_ih, wih_h, 2 * G3 * HID);
    k_cvt_f16<<<(2 * G3 * HID + 255) / 256, 256, 0, stream>>>(W_hh, whh_h, 2 * G3 * HID);
    k_cvt_f16<<<(LAT * HID + 255) / 256, 256, 0, stream>>>(W_out, wout_h, LAT * HID);

    // input projection
    int nx = BATCH * TLEN * HID;
    k_inproj<<<nx / 256, 256, 0, stream>>>(f0, loud, W_in, b_in, x0, nx);

    int mblocks = BATCH * TLEN / 16;   // 16384

    // layer 0
    k_xg_gemm<<<mblocks, 128, 0, stream>>>(x0, wih_h, b_ih, xgbuf);
    k_gru_scan<<<2, 128, 0, stream>>>(xgbuf, whh_h, b_hh, y0);
    // layer 1 (reuse xg buffer; y1 overwrites x0 buffer)
    k_xg_gemm<<<mblocks, 128, 0, stream>>>(y0, wih_h + G3 * HID, b_ih + G3, xgbuf);
    k_gru_scan<<<2, 128, 0, stream>>>(xgbuf, whh_h + G3 * HID, b_hh + G3, x0);
    // output projection + layernorm
    k_out_ln<<<mblocks, 128, 0, stream>>>(x0, wout_h, b_out, gamma, beta, out);
}
